// DocREModel1_66649302499935
// MI455X (gfx1250) — compile-verified, matrix operand order checked by reading, not run on hardware
//
#include <hip/hip_runtime.h>
#include <hip/hip_bf16.h>
#include <math.h>

// ---------- problem constants (from reference) ----------
constexpr int BS = 4, C = 1024, D = 768, H = 12, NE = 42, NM = 4;
constexpr int P  = NE * NE;        // 1764 pairs per doc
constexpr int PM = 1792;           // P padded to multiple of 32 for 2x2 WMMA blocking
constexpr int MT2 = PM / 32;       // 56  M macro-tiles (32 rows)
constexpr int NT2 = D / 32;        // 24  N macro-tiles (32 cols)
constexpr int CLS_OFFSET = 1;

// ---------- WMMA vector types ----------
typedef __attribute__((ext_vector_type(16))) __bf16 v16bf;
typedef __attribute__((ext_vector_type(8)))  __bf16 v8bf;
typedef __attribute__((ext_vector_type(8)))  float  v8f;

struct BfPair { v8bf lo, hi; };   // 32B, bit_cast-able to v16bf

// ---------- workspace layout (bytes, all 256B aligned) ----------
constexpr size_t SZ_SEQBT = (size_t)BS * D * C * 2;        // 6,291,456  (B^T bf16)
constexpr size_t OFF_SEQBT = 0;
constexpr size_t SZ_EEMB  = (size_t)BS * NE * D * 4;       //   516,096
constexpr size_t OFF_EEMB = OFF_SEQBT + SZ_SEQBT;
constexpr size_t SZ_EATT  = (size_t)BS * NE * H * C * 4;   // 8,257,536
constexpr size_t OFF_EATT = OFF_EEMB + SZ_EEMB;
constexpr size_t OFF_ABF  = OFF_EATT + SZ_EATT;            // bf16 ht_att [BS][PM][C]

// =====================================================================
// Kernel 1: sequence_output f32 [BS][C][D] -> bf16 TRANSPOSED [BS][D][C]
// Output-coalesced (contiguous over c); reads are L2-served.
// K becomes the contiguous axis for the WMMA B operand.
// =====================================================================
__global__ void k_seq_to_bf16T(const float* __restrict__ seq,
                               __bf16* __restrict__ seqbT) {
    long long t = (long long)blockIdx.x * blockDim.x + threadIdx.x;
    const long long total = (long long)BS * D * C;
    if (t >= total) return;
    int c = (int)(t % C);
    int d = (int)((t / C) % D);
    int b = (int)(t / ((long long)C * D));
    seqbT[t] = (__bf16)seq[((size_t)b * C + c) * D + d];
}

// =====================================================================
// Kernel 2: e_emb[b,e,d] = masked logsumexp over NM mentions
// =====================================================================
__global__ void k_e_emb(const float* __restrict__ seq,
                        const int* __restrict__ starts,
                        const unsigned char* __restrict__ mmask,
                        float* __restrict__ e_emb) {
    int be = blockIdx.x;               // b*NE + e
    int b = be / NE;

    int   idx[NM];
    bool  msk[NM];
    bool  any = false;
    #pragma unroll
    for (int m = 0; m < NM; ++m) {
        int s = starts[be * NM + m] + CLS_OFFSET;
        s = s < 0 ? 0 : (s > C - 1 ? C - 1 : s);
        idx[m] = s;
        msk[m] = mmask[be * NM + m] != 0;
        any |= msk[m];
    }
    const float* sb = seq + (size_t)b * C * D;
    for (int d = threadIdx.x; d < D; d += blockDim.x) {
        float out = 0.0f;
        if (any) {
            float mx = -INFINITY;
            float v[NM];
            #pragma unroll
            for (int m = 0; m < NM; ++m) {
                v[m] = sb[(size_t)idx[m] * D + d];
                if (msk[m]) mx = fmaxf(mx, v[m]);
            }
            float s = 0.0f;
            #pragma unroll
            for (int m = 0; m < NM; ++m)
                if (msk[m]) s += __expf(v[m] - mx);
            out = mx + __logf(s);
        }
        e_emb[(size_t)be * D + d] = out;
    }
}

// =====================================================================
// Kernel 3: e_att[b,e,h,c] = mean over masked mentions of attention[b,h,idx,c]
// =====================================================================
__global__ void k_e_att(const float* __restrict__ att,
                        const int* __restrict__ starts,
                        const unsigned char* __restrict__ mmask,
                        float* __restrict__ e_att) {
    long long t = (long long)blockIdx.x * blockDim.x + threadIdx.x;
    const long long total = (long long)BS * NE * H * C;
    if (t >= total) return;
    int c = (int)(t % C);
    int h = (int)((t / C) % H);
    int e = (int)((t / ((long long)C * H)) % NE);
    int b = (int)(t / ((long long)C * H * NE));
    int be = b * NE + e;

    float acc = 0.0f;
    int   cnt = 0;
    #pragma unroll
    for (int m = 0; m < NM; ++m) {
        if (mmask[be * NM + m]) {
            int s = starts[be * NM + m] + CLS_OFFSET;
            s = s < 0 ? 0 : (s > C - 1 ? C - 1 : s);
            acc += att[(((size_t)b * H + h) * C + s) * C + c];
            ++cnt;
        }
    }
    e_att[t] = (cnt > 0) ? acc / (float)cnt : 0.0f;
}

// =====================================================================
// Kernel 4: per pair (b,p):
//   hs/ts copies -> d_out slabs 0/1
//   ht_att[c] = mean_h(e_att[hi]*e_att[ti]); normalize; -> bf16 A row
// =====================================================================
__global__ void k_pairs(const float* __restrict__ e_emb,
                        const float* __restrict__ e_att,
                        const int*   __restrict__ hts,
                        float*  __restrict__ out,      // [3][BS*P][D]
                        __bf16* __restrict__ Abf) {    // [BS][PM][C]
    int bp = blockIdx.x;
    int b = bp / PM;
    int p = bp % PM;
    int tid = threadIdx.x;
    __bf16* arow = Abf + ((size_t)b * PM + p) * C;

    if (p >= P) {                       // zero pad rows every call
        for (int c = tid; c < C; c += 256) arow[c] = (__bf16)0.0f;
        return;
    }

    int hi = hts[((size_t)b * P + p) * 2 + 0];
    int ti = hts[((size_t)b * P + p) * 2 + 1];

    // hs / ts gathers
    const float* eh = e_emb + ((size_t)b * NE + hi) * D;
    const float* et = e_emb + ((size_t)b * NE + ti) * D;
    float* o0 = out + ((size_t)b * P + p) * D;                       // slab 0
    float* o1 = out + (size_t)BS * P * D + ((size_t)b * P + p) * D;  // slab 1
    for (int d = tid; d < D; d += 256) { o0[d] = eh[d]; o1[d] = et[d]; }

    // ht_att over c, mean over heads
    const float* ah = e_att + ((size_t)b * NE + hi) * H * C;
    const float* at = e_att + ((size_t)b * NE + ti) * H * C;
    float vals[C / 256];
    float partial = 0.0f;
    #pragma unroll
    for (int j = 0; j < C / 256; ++j) {
        int c = j * 256 + tid;
        float acc = 0.0f;
        #pragma unroll
        for (int h = 0; h < H; ++h)
            acc += ah[(size_t)h * C + c] * at[(size_t)h * C + c];
        acc *= (1.0f / (float)H);
        vals[j] = acc;
        partial += acc;
    }
    __shared__ float red[256];
    red[tid] = partial;
    __syncthreads();
    for (int s = 128; s > 0; s >>= 1) {
        if (tid < s) red[tid] += red[tid + s];
        __syncthreads();
    }
    float inv = 1.0f / (red[0] + 1e-5f);
    #pragma unroll
    for (int j = 0; j < C / 256; ++j) {
        int c = j * 256 + tid;
        arow[c] = (__bf16)(vals[j] * inv);
    }
}

// =====================================================================
// Kernel 5: rs[b] = A[b] (PM x C) @ B^T[b] (D x C), both bf16 K-contig.
// One wave computes a 32x32 output tile via 2x2 WMMA register blocking:
// per K-step -> 2 A-frags + 2 B-frags (8x b128) feed 4 independent WMMAs.
// =====================================================================
__device__ __forceinline__ v16bf load_frag(const __bf16* __restrict__ p) {
    BfPair t;
    t.lo = *(const v8bf*)(p);        // K octet 0 (per half)
    t.hi = *(const v8bf*)(p + 16);   // K octet +16 (per half)
    return __builtin_bit_cast(v16bf, t);
}

__device__ __forceinline__ void store_tile(float* __restrict__ outRS,
                                           int b, int row0, int col,
                                           const v8f& acc) {
    float* orow = outRS + (size_t)b * P * D + (size_t)row0 * D + col;
    if (row0 + 7 < P) {                      // fast path: unguarded stores
        #pragma unroll
        for (int i = 0; i < 8; ++i)
            orow[(size_t)i * D] = acc[i];
    } else {                                 // only the last M macro-tile
        #pragma unroll
        for (int i = 0; i < 8; ++i)
            if (row0 + i < P) orow[(size_t)i * D] = acc[i];
    }
}

__global__ void k_gemm_wmma(const __bf16* __restrict__ A,
                            const __bf16* __restrict__ BT,
                            float* __restrict__ outRS) {  // slab 2 base
    const int lane = threadIdx.x & 31;
    const int wave = threadIdx.x >> 5;
    int tile = blockIdx.x * 8 + wave;        // BS*MT2*NT2 = 5376, multiple of 8
    int b  = tile / (MT2 * NT2);
    int r  = tile % (MT2 * NT2);
    int mt = r / NT2;                        // 32-row macro tile
    int nt = r % NT2;                        // 32-col macro tile

    const int rr   = lane & 15;
    const int half = lane >> 4;              // K-octet select {0-7,16-23} vs {8-15,24-31}

    const __bf16* ap0 = A  + ((size_t)b * PM + (mt * 32 + rr)) * C + half * 8;
    const __bf16* ap1 = ap0 + (size_t)16 * C;
    const __bf16* bp0 = BT + ((size_t)b * D  + (nt * 32 + rr)) * C + half * 8;
    const __bf16* bp1 = bp0 + (size_t)16 * C;

    v8f acc00 = {}, acc01 = {}, acc10 = {}, acc11 = {};
    #pragma unroll 2
    for (int k0 = 0; k0 < C; k0 += 32) {
        v16bf a0 = load_frag(ap0 + k0);
        v16bf a1 = load_frag(ap1 + k0);
        v16bf b0 = load_frag(bp0 + k0);
        v16bf b1 = load_frag(bp1 + k0);
        acc00 = __builtin_amdgcn_wmma_f32_16x16x32_bf16(
                    false, a0, false, b0, (short)0, acc00, false, false);
        acc01 = __builtin_amdgcn_wmma_f32_16x16x32_bf16(
                    false, a0, false, b1, (short)0, acc01, false, false);
        acc10 = __builtin_amdgcn_wmma_f32_16x16x32_bf16(
                    false, a1, false, b0, (short)0, acc10, false, false);
        acc11 = __builtin_amdgcn_wmma_f32_16x16x32_bf16(
                    false, a1, false, b1, (short)0, acc11, false, false);
    }

    // D layout per 16x16 tile: VGPR i -> M = i + 8*half, N = rr
    const int m0 = mt * 32 + 8 * half;
    const int n0 = nt * 32 + rr;
    store_tile(outRS, b, m0,      n0,      acc00);
    store_tile(outRS, b, m0,      n0 + 16, acc01);
    store_tile(outRS, b, m0 + 16, n0,      acc10);
    store_tile(outRS, b, m0 + 16, n0 + 16, acc11);
}

// =====================================================================
extern "C" void kernel_launch(void* const* d_in, const int* in_sizes, int n_in,
                              void* d_out, int out_size, void* d_ws, size_t ws_size,
                              hipStream_t stream) {
    const float*         seq    = (const float*)d_in[0];          // [BS,C,D]
    const float*         att    = (const float*)d_in[1];          // [BS,H,C,C]
    const int*           starts = (const int*)d_in[2];            // [BS,NE,NM]
    const unsigned char* mmask  = (const unsigned char*)d_in[3];  // [BS,NE,NM] bool
    const int*           hts    = (const int*)d_in[4];            // [BS,P,2]
    float* out = (float*)d_out;                                   // [3, BS*P, D]

    char* ws = (char*)d_ws;
    __bf16* seqbT = (__bf16*)(ws + OFF_SEQBT);
    float*  e_emb = (float*)(ws + OFF_EEMB);
    float*  e_att = (float*)(ws + OFF_EATT);
    __bf16* Abf   = (__bf16*)(ws + OFF_ABF);

    // 1) seq -> bf16, transposed so K is contiguous for WMMA B fragments
    {
        long long n = (long long)BS * D * C;
        k_seq_to_bf16T<<<(int)((n + 255) / 256), 256, 0, stream>>>(seq, seqbT);
    }
    // 2) e_emb (logsumexp pooling)
    k_e_emb<<<BS * NE, 256, 0, stream>>>(seq, starts, mmask, e_emb);
    // 3) e_att (attention gather + mention mean)
    {
        long long total = (long long)BS * NE * H * C;
        k_e_att<<<(int)((total + 255) / 256), 256, 0, stream>>>(att, starts, mmask, e_att);
    }
    // 4) pair stage: hs/ts + normalized ht_att as bf16 A-matrix
    k_pairs<<<BS * PM, 256, 0, stream>>>(e_emb, e_att, hts, out, Abf);
    // 5) rs GEMM via WMMA bf16 -> f32 (2x2 blocked), written to slab 2
    {
        float* rsOut = out + (size_t)2 * BS * P * D;
        int blocks = (BS * MT2 * NT2) / 8;   // 672, exact
        k_gemm_wmma<<<blocks, 256, 0, stream>>>(Abf, seqbT, rsOut);
    }
}
